// HPHA_45311904973052
// MI455X (gfx1250) — matrix operand order repaired and checked by reading, not compiled, not run on hardware
//
#include <hip/hip_runtime.h>
#include <hip/hip_bf16.h>
#include <math.h>

// Problem constants (match reference)
#define BB 2
#define LL 5
#define NN 10          // B*L
#define CC 64
#define HH 128
#define WW 256
#define PP 32768       // H*W = 2^15
#define THRESH 0.01f

typedef __attribute__((ext_vector_type(16))) _Float16 v16h;
typedef __attribute__((ext_vector_type(8)))  float    v8f;

// Async global->LDS DMA operand types (match builtin signature exactly:
// pointee = int __attribute__((vector_size(16))), AS1 for global, AS3 for LDS)
typedef int v4i __attribute__((vector_size(4 * sizeof(int))));
typedef __attribute__((address_space(1))) v4i* gp_v4i;
typedef __attribute__((address_space(3))) v4i* lp_v4i;

// 5x5 Gaussian taps, sigma=1: 1/(2*pi) * exp(-(dx^2+dy^2)/2), row-major dy,dx.
__device__ __constant__ float GK25[25] = {
  0.0029150245f, 0.0130642333f, 0.0215392793f, 0.0130642333f, 0.0029150245f,
  0.0130642333f, 0.0585498315f, 0.0965323526f, 0.0585498315f, 0.0130642333f,
  0.0215392793f, 0.0965323526f, 0.1591549431f, 0.0965323526f, 0.0215392793f,
  0.0130642333f, 0.0585498315f, 0.0965323526f, 0.0585498315f, 0.0130642333f,
  0.0029150245f, 0.0130642333f, 0.0215392793f, 0.0130642333f, 0.0029150245f,
};

// ---------------------------------------------------------------------------
// Kernel 1: conf[n,p] = max(sigmoid(psm[n,0,p]), sigmoid(psm[n,1,p])) * enc[n]
// ---------------------------------------------------------------------------
__global__ void k_conf(const float* __restrict__ psm,
                       const float* __restrict__ inv_delay,
                       const float* __restrict__ ewc_w,
                       const float* __restrict__ ewc_b,
                       float* __restrict__ conf) {
  int i = blockIdx.x * blockDim.x + threadIdx.x;   // 0 .. N*P-1
  int n = i >> 15;
  int p = i & (PP - 1);
  float a = psm[(size_t)n * 2 * PP + p];
  float b = psm[(size_t)n * 2 * PP + PP + p];
  float sa = 1.f / (1.f + __expf(-a));
  float sb = 1.f / (1.f + __expf(-b));
  float enc = tanhf(inv_delay[n] * ewc_w[0] + ewc_b[0]) + 1.f;
  conf[i] = fmaxf(sa, sb) * enc;
}

// ---------------------------------------------------------------------------
// Kernel 2: 5x5 Gaussian smooth as im2col WMMA (16 px x 32 taps) -> mask.
// One wave handles 16 consecutive pixels along W for fixed (n,h).
// comm-rate numerator: LDS block reduction (ds_add_f32) then ONE global
// atomic per block (2560 total) instead of 40960 same-address atomics.
// ---------------------------------------------------------------------------
__global__ void k_mask(const float* __restrict__ conf,
                       float* __restrict__ mask,
                       float* __restrict__ commSum) {
  __shared__ float blockSum;
  if (threadIdx.x == 0) blockSum = 0.f;
  __syncthreads();

  const int gtid = blockIdx.x * blockDim.x + threadIdx.x;
  const int seg  = gtid >> 5;              // 0 .. N*H*(W/16)-1
  const int ln   = threadIdx.x & 31;       // wave32 lane
  const int wsegs = WW / 16;               // 16
  const int n  = seg / (HH * wsegs);
  const int r  = seg % (HH * wsegs);
  const int h  = r / wsegs;
  const int w0 = (r % wsegs) * 16;

  const int hh = ln >> 4;                  // lane half selects K subset
  const int m  = ln & 15;                  // output pixel row (A-matrix M)
  const float* cn = conf + (size_t)n * PP;

  // A (16x32 f16): A[m][k] = conf at tap k (k = dy*5+dx, 25 taps, rest zero).
  // 16-bit A layout: lane<16 -> K = e+8*hh for e<8, e+8+8*hh for e>=8.
  v16h a;
#pragma unroll
  for (int e = 0; e < 16; ++e) {
    int k = (e < 8) ? (e + 8 * hh) : (e + 8 + 8 * hh);
    float v = 0.f;
    if (k < 25) {
      int dy = k / 5, dx = k - 5 * dy;
      int gy = h - 2 + dy;
      int gx = w0 + m - 2 + dx;
      if (gy >= 0 && gy < HH && gx >= 0 && gx < WW) v = cn[gy * WW + gx];
    }
    a[e] = (_Float16)v;
  }

  // B (32x16 f16): Gaussian taps replicated into every column.
  // 16-bit B layout (same K order as A): K = e+8*hh for e<8, e+8+8*hh for e>=8.
  v16h bm;
#pragma unroll
  for (int e = 0; e < 16; ++e) {
    int k = (e < 8) ? (e + 8 * hh) : (e + 8 + 8 * hh);
    bm[e] = (_Float16)((k < 25) ? GK25[k] : 0.f);
  }

  v8f c = {};
  v8f d = __builtin_amdgcn_wmma_f32_16x16x32_f16(
      /*neg_a=*/false, a, /*neg_b=*/false, bm,
      /*c_mod=*/(short)0, c, /*reuse_a=*/false, /*reuse_b=*/false);

  // D (16x16 f32): lanes 0..15 hold M=r (element r), lanes 16..31 hold M=8+r.
  // All columns identical -> lanes 0 and 16 write the 16 smoothed pixels.
  if ((ln & 15) == 0) {
    const int mbase = hh * 8;
    float* mp = mask + (size_t)n * PP + h * WW + w0 + mbase;
    const bool ego = (n % LL) == 0;
    float local = 0.f;
#pragma unroll
    for (int q = 0; q < 8; ++q) {
      float mv = (d[q] > THRESH) ? 1.f : 0.f;
      mp[q] = mv;
      local += ego ? 1.f : mv;
    }
    atomicAdd(&blockSum, local);           // ds_add_f32, 16 contributors
  }
  __syncthreads();
  if (threadIdx.x == 0) atomicAdd(commSum, blockSum);
}

// ---------------------------------------------------------------------------
// Kernel 3: per-pixel cross-agent attention. One thread per (b, pixel).
// Pass 1 reads all 5 CAV features (ego cached in regs), pass 2 re-reads the
// 4 non-ego CAVs (L2-resident). Writes x_fuse straight into d_out.
// ---------------------------------------------------------------------------
__global__ void k_fuse(const float* __restrict__ x,
                       const float* __restrict__ mask,
                       const float* __restrict__ inv_delay,
                       const float* __restrict__ ew_w,
                       const float* __restrict__ ew_b,
                       float* __restrict__ xfuse) {
  const int gid = blockIdx.x * blockDim.x + threadIdx.x;  // 0 .. B*P-1
  const int b = gid >> 15;
  const int p = gid & (PP - 1);
  const int n0 = b * LL;
  const float eww = ew_w[0], ewb = ew_b[0];

  float g[LL];  // combined enw * mask scale per cav
#pragma unroll
  for (int l = 0; l < LL; ++l) {
    float enw = tanhf(inv_delay[n0 + l] * eww + ewb) + 1.f;
    float mk  = (l == 0) ? 1.f : mask[(size_t)(n0 + l) * PP + p];
    g[l] = enw * mk;
  }

  const float* xb = x + (size_t)n0 * CC * PP + p;
  float ego[CC];
  float s0 = 0.f, s1 = 0.f, s2 = 0.f, s3 = 0.f, s4 = 0.f;
#pragma unroll
  for (int c = 0; c < CC; ++c) {
    const float* xc = xb + (size_t)c * PP;
    float e = xc[0] * g[0];
    ego[c] = e;
    s0 += e * e;
    s1 += e * (xc[(size_t)1 * CC * PP] * g[1]);
    s2 += e * (xc[(size_t)2 * CC * PP] * g[2]);
    s3 += e * (xc[(size_t)3 * CC * PP] * g[3]);
    s4 += e * (xc[(size_t)4 * CC * PP] * g[4]);
  }
  const float scale = 0.125f;  // 1/sqrt(64)
  s0 *= scale; s1 *= scale; s2 *= scale; s3 *= scale; s4 *= scale;
  float mx = fmaxf(fmaxf(fmaxf(s0, s1), fmaxf(s2, s3)), s4);
  float w0 = __expf(s0 - mx), w1 = __expf(s1 - mx), w2 = __expf(s2 - mx),
        w3 = __expf(s3 - mx), w4 = __expf(s4 - mx);
  float inv = 1.f / (w0 + w1 + w2 + w3 + w4);
  w0 *= inv; w1 *= inv; w2 *= inv; w3 *= inv; w4 *= inv;

  float* of = xfuse + (size_t)b * CC * PP + p;
#pragma unroll
  for (int c = 0; c < CC; ++c) {
    const float* xc = xb + (size_t)c * PP;
    float f = w0 * ego[c]
            + w1 * (xc[(size_t)1 * CC * PP] * g[1])
            + w2 * (xc[(size_t)2 * CC * PP] * g[2])
            + w3 * (xc[(size_t)3 * CC * PP] * g[3])
            + w4 * (xc[(size_t)4 * CC * PP] * g[4]);
    of[(size_t)c * PP] = f;
  }
}

// ---------------------------------------------------------------------------
// Kernel 4: per-(b,c) sum/max reduction over 32768 pixels.
// Streams data LDS-staged via gfx1250 async global->LDS DMA (double buffered,
// per-wave ASYNCcnt ordering; each lane reads back only the 16B it staged,
// so no cross-lane barrier is needed). Falls back to plain loads if the
// async builtins are not declared by this toolchain.
// ---------------------------------------------------------------------------
#if __has_builtin(__builtin_amdgcn_global_load_async_to_lds_b128) && \
    __has_builtin(__builtin_amdgcn_s_wait_asynccnt)
#define USE_ASYNC_LDS 1
#else
#define USE_ASYNC_LDS 0
#endif

__global__ void k_pool(const float* __restrict__ xfuse,
                       float* __restrict__ psum, float* __restrict__ pmax) {
  const int bc = blockIdx.x;  // 0..127
  const float* src = xfuse + (size_t)bc * PP;
  const int tid = threadIdx.x;
  float s = 0.f, mx = -INFINITY;

#if USE_ASYNC_LDS
  __shared__ __align__(16) float stage[2][256 * 4];   // 2 x 4KB double buffer
  const int CHUNK = 256 * 4;                          // floats per chunk
  const int NCHUNK = PP / CHUNK;                      // 32
  // prologue: stage chunk 0
  __builtin_amdgcn_global_load_async_to_lds_b128(
      (gp_v4i)(src + tid * 4),
      (lp_v4i)&stage[0][tid * 4], 0, 0);
  for (int ch = 0; ch < NCHUNK; ++ch) {
    const int cur = ch & 1;
    if (ch + 1 < NCHUNK) {
      __builtin_amdgcn_global_load_async_to_lds_b128(
          (gp_v4i)(src + (ch + 1) * CHUNK + tid * 4),
          (lp_v4i)&stage[cur ^ 1][tid * 4], 0, 0);
      __builtin_amdgcn_s_wait_asynccnt(1);            // chunk `ch` landed
    } else {
      __builtin_amdgcn_s_wait_asynccnt(0);
    }
    asm volatile("" ::: "memory");                    // keep LDS reads below wait
    float4 v = *(const float4*)&stage[cur][tid * 4];
    s += v.x + v.y + v.z + v.w;
    mx = fmaxf(mx, fmaxf(fmaxf(v.x, v.y), fmaxf(v.z, v.w)));
  }
#else
  for (int i = tid; i < PP; i += blockDim.x) {
    float v = src[i];
    s += v;
    mx = fmaxf(mx, v);
  }
#endif

  __shared__ float ss[8], sm[8];
#pragma unroll
  for (int off = 16; off > 0; off >>= 1) {
    s += __shfl_down(s, off, 32);
    mx = fmaxf(mx, __shfl_down(mx, off, 32));
  }
  const int wv = tid >> 5;
  if ((tid & 31) == 0) { ss[wv] = s; sm[wv] = mx; }
  __syncthreads();
  if (tid == 0) {
    float S = 0.f, M = -INFINITY;
    for (int i = 0; i < (int)(blockDim.x >> 5); ++i) {
      S += ss[i];
      M = fmaxf(M, sm[i]);
    }
    psum[bc] = S;
    pmax[bc] = M;
  }
}

// ---------------------------------------------------------------------------
// Kernel 5: channel gate = sigmoid(fc(avg) + fc(max)), fc = relu(v W1^T) W2^T
// ---------------------------------------------------------------------------
__global__ void k_gate(const float* __restrict__ psum,
                       const float* __restrict__ pmax,
                       const float* __restrict__ w1,   // (4,64)
                       const float* __restrict__ w2,   // (64,4)
                       float* __restrict__ gate) {
  const int b = blockIdx.x;   // 0..1
  const int c = threadIdx.x;  // 0..63
  const float invP = 1.f / (float)PP;
  float acc = 0.f;
#pragma unroll
  for (int j = 0; j < 4; ++j) {
    float ha = 0.f, hm = 0.f;
    for (int k = 0; k < CC; ++k) {
      float w = w1[j * CC + k];
      ha += (psum[b * CC + k] * invP) * w;
      hm += pmax[b * CC + k] * w;
    }
    ha = fmaxf(ha, 0.f);
    hm = fmaxf(hm, 0.f);
    acc += (ha + hm) * w2[c * 4 + j];
  }
  gate[b * CC + c] = 1.f / (1.f + __expf(-acc));
}

// ---------------------------------------------------------------------------
// Kernel 6: out *= gate (in place on d_out); thread 0 also writes comm_rate.
// ---------------------------------------------------------------------------
__global__ void k_out(float* __restrict__ out,
                      const float* __restrict__ gate,
                      const float* __restrict__ commSum) {
  const size_t i = (size_t)blockIdx.x * blockDim.x + threadIdx.x;
  const int bc = (int)(i >> 15);
  out[i] = out[i] * gate[bc];
  if (i == 0) out[(size_t)BB * CC * PP] = commSum[0] * (1.f / (float)(NN * PP));
}

// ---------------------------------------------------------------------------
extern "C" void kernel_launch(void* const* d_in, const int* in_sizes, int n_in,
                              void* d_out, int out_size, void* d_ws, size_t ws_size,
                              hipStream_t stream) {
  const float* x     = (const float*)d_in[0];
  const float* psm   = (const float*)d_in[1];
  const float* invd  = (const float*)d_in[2];
  const float* ew_w  = (const float*)d_in[3];
  const float* ew_b  = (const float*)d_in[4];
  const float* ewc_w = (const float*)d_in[5];
  const float* ewc_b = (const float*)d_in[6];
  const float* w1    = (const float*)d_in[7];
  const float* w2    = (const float*)d_in[8];
  float* out = (float*)d_out;

  float* ws   = (float*)d_ws;
  float* conf = ws;                         // N*P floats
  float* mask = conf + (size_t)NN * PP;     // N*P floats
  float* psum = mask + (size_t)NN * PP;     // 128
  float* pmax = psum + BB * CC;             // 128
  float* gate = pmax + BB * CC;             // 128
  float* csum = gate + BB * CC;             // 1

  (void)hipMemsetAsync(csum, 0, sizeof(float), stream);

  k_conf<<<(NN * PP) / 256, 256, 0, stream>>>(psm, invd, ewc_w, ewc_b, conf);
  k_mask<<<((NN * PP) / 16) * 32 / 256, 256, 0, stream>>>(conf, mask, csum);
  k_fuse<<<(BB * PP) / 256, 256, 0, stream>>>(x, mask, invd, ew_w, ew_b, out);
  k_pool<<<BB * CC, 256, 0, stream>>>(out, psum, pmax);
  k_gate<<<BB, CC, 0, stream>>>(psum, pmax, w1, w2, gate);
  k_out<<<((size_t)BB * CC * PP) / 256, 256, 0, stream>>>(out, gate, csum);
}
